// HeteroGATEncoder_28467043238289
// MI455X (gfx1250) — matrix operand order repaired and checked by reading, not compiled, not run on hardware
//
#include <hip/hip_runtime.h>

typedef __attribute__((ext_vector_type(16))) _Float16 v16h;
typedef __attribute__((ext_vector_type(8)))  _Float16 v8h;
typedef __attribute__((ext_vector_type(8)))  float    v8f;

#define NEG_SLOPE 0.2f

__device__ __forceinline__ float atomicMaxF32(float* addr, float val) {
  // int compare == float compare for val >= 0; reversed unsigned compare for val < 0.
  // Buffer must be initialized to -inf (0xFF800000): works for both branches.
  if (val >= 0.0f)
    return __int_as_float(atomicMax((int*)addr, __float_as_int(val)));
  else
    return __uint_as_float(atomicMin((unsigned int*)addr, __float_as_uint(val)));
}

extern "C" __global__ void k_cvt_f16(const float* __restrict__ s, _Float16* __restrict__ d, int n) {
  int i = blockIdx.x * blockDim.x + threadIdx.x;
  if (i < n) d[i] = (_Float16)s[i];
}

// d[N,K] = (f16) transpose of s[K,N]
extern "C" __global__ void k_cvt_t(const float* __restrict__ s, _Float16* __restrict__ d,
                                   int K, int N) {
  int i = blockIdx.x * blockDim.x + threadIdx.x;
  if (i >= K * N) return;
  int k = i / N, n = i % N;
  d[(size_t)n * K + k] = (_Float16)s[i];
}

extern "C" __global__ void k_fill(float* __restrict__ p, float v, int n) {
  int i = blockIdx.x * blockDim.x + threadIdx.x;
  if (i < n) p[i] = v;
}

// C[M,N] = A[M,K](f16, row-major) @ Bt[N,K](f16, pre-transposed) [+bias] [+relu]
// One wave32 computes a 16x64 output tile: 4 WMMA accumulators sharing one A fragment.
// Requires M%16==0, K%32==0, N%64==0.
extern "C" __global__ void k_gemm_wmma(const _Float16* __restrict__ A,
                                       const _Float16* __restrict__ Bt,
                                       const float* __restrict__ bias,
                                       float* __restrict__ C,
                                       int M, int N, int K, int act) {
  const int lane = threadIdx.x & 31;
  const int wid  = blockIdx.x * (blockDim.x >> 5) + (threadIdx.x >> 5); // wave-uniform
  const int Nt   = N >> 6;
  const int tiles = (M >> 4) * Nt;
  if (wid >= tiles) return;            // whole wave exits together -> EXEC all-1s at WMMA
  const int mt   = wid / Nt, nt = wid % Nt;
  const int half = lane >> 4;          // 0: lanes 0-15, 1: lanes 16-31
  const int l16  = lane & 15;
  const size_t arow = (size_t)(mt * 16 + l16) * K;
  const size_t brow = (size_t)(nt * 64 + l16) * K; // Bt row for column tile 0

  v8f acc0 = {}, acc1 = {}, acc2 = {}, acc3 = {};
  for (int kk = 0; kk < K; kk += 32) {
    // A 16x32 fragment: 16 contiguous halves per lane (k = 16*half + s), matching B
    union { v16h v; v8h h[2]; } ua;
    ua.h[0] = *(const v8h*)(A + arow + kk + half * 16);
    ua.h[1] = *(const v8h*)(A + arow + kk + half * 16 + 8);
    // B 32x16 fragments: 16 contiguous halves (32B) per lane per tile from Bt
    const _Float16* bp = Bt + brow + kk + half * 16;
    v16h b0 = *(const v16h*)(bp);
    v16h b1 = *(const v16h*)(bp + (size_t)16 * K);
    v16h b2 = *(const v16h*)(bp + (size_t)32 * K);
    v16h b3 = *(const v16h*)(bp + (size_t)48 * K);
    acc0 = __builtin_amdgcn_wmma_f32_16x16x32_f16(false, ua.v, false, b0, (short)0, acc0, false, false);
    acc1 = __builtin_amdgcn_wmma_f32_16x16x32_f16(false, ua.v, false, b1, (short)0, acc1, false, false);
    acc2 = __builtin_amdgcn_wmma_f32_16x16x32_f16(false, ua.v, false, b2, (short)0, acc2, false, false);
    acc3 = __builtin_amdgcn_wmma_f32_16x16x32_f16(false, ua.v, false, b3, (short)0, acc3, false, false);
  }
  const int nc = nt * 64 + l16;
  float bv0 = 0.f, bv1 = 0.f, bv2 = 0.f, bv3 = 0.f;
  if (bias) { bv0 = bias[nc]; bv1 = bias[nc + 16]; bv2 = bias[nc + 32]; bv3 = bias[nc + 48]; }
#pragma unroll
  for (int r = 0; r < 8; ++r) {
    const size_t row = (size_t)(mt * 16 + r + 8 * half) * N + nc;
    float v0 = acc0[r] + bv0, v1 = acc1[r] + bv1, v2 = acc2[r] + bv2, v3 = acc3[r] + bv3;
    if (act) {
      v0 = v0 > 0.f ? v0 : 0.f; v1 = v1 > 0.f ? v1 : 0.f;
      v2 = v2 > 0.f ? v2 : 0.f; v3 = v3 > 0.f ? v3 : 0.f;
    }
    C[row] = v0; C[row + 16] = v1; C[row + 32] = v2; C[row + 48] = v3;
  }
}

// out[n,h] = sum_c h[n, h*ch + c] * avec[h*ch + c]
extern "C" __global__ void k_logits(const float* __restrict__ hmat, const float* __restrict__ avec,
                                    float* __restrict__ o, int n, int heads, int ch) {
  int i = blockIdx.x * blockDim.x + threadIdx.x;
  if (i >= n * heads) return;
  int node = i / heads, h = i % heads;
  const float* hp = hmat + (size_t)node * heads * ch + h * ch;
  const float* ap = avec + h * ch;
  float s = 0.0f;
  for (int c = 0; c < ch; ++c) s += hp[c] * ap[c];
  o[i] = s;
}

// e = leaky_relu(al_s[src] + al_d[dst]); segment max into mb
extern "C" __global__ void k_edge_a(const float* __restrict__ als, const float* __restrict__ ald,
                                    const int* __restrict__ src, const int* __restrict__ dst,
                                    float* __restrict__ eb, float* __restrict__ mb,
                                    int E, int heads) {
  int i = blockIdx.x * blockDim.x + threadIdx.x;
  if (i >= E * heads) return;
  int e = i / heads, h = i % heads;
  float v = als[src[e] * heads + h] + ald[dst[e] * heads + h];
  v = (v > 0.0f) ? v : NEG_SLOPE * v;
  eb[i] = v;
  atomicMaxF32(&mb[dst[e] * heads + h], v);
}

// ex = exp(e - m[dst]); segment sum into zb; eb overwritten with ex
extern "C" __global__ void k_edge_b(float* __restrict__ eb, const float* __restrict__ mb,
                                    const int* __restrict__ dst, float* __restrict__ zb,
                                    int E, int heads) {
  int i = blockIdx.x * blockDim.x + threadIdx.x;
  if (i >= E * heads) return;
  int e = i / heads, h = i % heads;
  float ex = __expf(eb[i] - mb[dst[e] * heads + h]);
  eb[i] = ex;
  atomicAdd(&zb[dst[e] * heads + h], ex);
}

// acc[dst, :] += h_src[src, :] * (ex / (z[dst] + 1e-16)); 4 channels per thread
extern "C" __global__ void k_edge_c(const float* __restrict__ hsrc, const float* __restrict__ eb,
                                    const float* __restrict__ zb,
                                    const int* __restrict__ src, const int* __restrict__ dst,
                                    float* __restrict__ acc, int E, int heads, int ch) {
  int i = blockIdx.x * blockDim.x + threadIdx.x;
  const int W  = heads * ch;
  const int W4 = W >> 2;
  if (i >= E * W4) return;
  int e = i / W4, c = (i % W4) << 2, h = c / ch;   // 4 channels share one head (ch=64)
  int s = src[e], d4 = dst[e];
  float alpha = eb[e * heads + h] / (zb[d4 * heads + h] + 1e-16f);
  const float4 hv = *(const float4*)(hsrc + (size_t)s * W + c);
  float* ap = acc + (size_t)d4 * W + c;
  atomicAdd(ap + 0, hv.x * alpha);
  atomicAdd(ap + 1, hv.y * alpha);
  atomicAdd(ap + 2, hv.z * alpha);
  atomicAdd(ap + 3, hv.w * alpha);
}

// o = [elu](acc + b1 [+ b2])
extern "C" __global__ void k_finalize(float* __restrict__ o, const float* __restrict__ acc,
                                      const float* __restrict__ b1, const float* __restrict__ b2,
                                      int n, int w, int do_elu) {
  int i = blockIdx.x * blockDim.x + threadIdx.x;
  if (i >= n * w) return;
  int c = i % w;
  float v = acc[i] + b1[c] + (b2 ? b2[c] : 0.0f);
  if (do_elu) v = v > 0.0f ? v : (__expf(v) - 1.0f);
  o[i] = v;
}

extern "C" void kernel_launch(void* const* d_in, const int* in_sizes, int n_in,
                              void* d_out, int out_size, void* d_ws, size_t ws_size,
                              hipStream_t stream) {
  (void)in_sizes; (void)n_in; (void)out_size; (void)ws_size;
  const int ND = 20000, NP = 40000, E = 300000;
  const int H = 4, Cc = 64, HID = 256, OUT = 64, IND = 128, INP = 256;

  const float* x_drug    = (const float*)d_in[0];
  const float* x_prot    = (const float*)d_in[1];
  const int*   src_dp    = (const int*)d_in[2];
  const int*   dst_dp    = (const int*)d_in[3];
  const int*   src_pd    = (const int*)d_in[4];
  const int*   dst_pd    = (const int*)d_in[5];
  const int*   src_pp    = (const int*)d_in[6];
  const int*   dst_pp    = (const int*)d_in[7];
  const float* linW_drug = (const float*)d_in[8];
  const float* linb_drug = (const float*)d_in[9];
  const float* linW_prot = (const float*)d_in[10];
  const float* linb_prot = (const float*)d_in[11];
  const float* W_hid     = (const float*)d_in[12];
  const float* asrc_hid  = (const float*)d_in[13];
  const float* adst_hid  = (const float*)d_in[14];
  const float* b_hid     = (const float*)d_in[15];
  const float* W_out     = (const float*)d_in[16];
  const float* asrc_out  = (const float*)d_in[17];
  const float* adst_out  = (const float*)d_in[18];
  const float* b_out     = (const float*)d_in[19];

  // ---- workspace carve-out ----
  char* base = (char*)d_ws;
  size_t off = 0;
  auto carve = [&](size_t bytes) -> void* {
    off = (off + 255) & ~(size_t)255;
    void* p = base + off;
    off += bytes;
    return p;
  };
  float* xd   = (float*)carve((size_t)ND * HID * 4);
  float* xp   = (float*)carve((size_t)NP * HID * 4);
  float* hd   = (float*)carve((size_t)ND * HID * 4);
  float* hp   = (float*)carve((size_t)NP * HID * 4);
  float* accd = (float*)carve((size_t)ND * HID * 4);
  float* accp = (float*)carve((size_t)NP * HID * 4);
  float* als  = (float*)carve((size_t)NP * H * 4);
  float* ald  = (float*)carve((size_t)NP * H * 4);
  float* mb   = (float*)carve((size_t)NP * H * 4);
  float* zb   = (float*)carve((size_t)NP * H * 4);
  float* eb   = (float*)carve((size_t)E * H * 4);
  _Float16* xdh  = (_Float16*)carve((size_t)ND * HID * 2);
  _Float16* xph  = (_Float16*)carve((size_t)NP * HID * 2);
  _Float16* wld  = (_Float16*)carve((size_t)IND * HID * 2);   // transposed [HID,IND]
  _Float16* wlp  = (_Float16*)carve((size_t)INP * HID * 2);   // transposed [HID,INP]
  _Float16* whid = (_Float16*)carve((size_t)2 * 3 * HID * HID * 2); // each transposed
  _Float16* wout = (_Float16*)carve((size_t)3 * HID * OUT * 2);     // each transposed [OUT,HID]

  const float NEG_INF = -__builtin_inff();

  // ---- launch helpers ----
  auto cvt = [&](const float* s, _Float16* d, int n) {
    k_cvt_f16<<<(n + 255) / 256, 256, 0, stream>>>(s, d, n);
  };
  auto cvt_t = [&](const float* s, _Float16* d, int K, int N) {
    int t = K * N;
    k_cvt_t<<<(t + 255) / 256, 256, 0, stream>>>(s, d, K, N);
  };
  auto fill = [&](float* p, float v, int n) {
    k_fill<<<(n + 255) / 256, 256, 0, stream>>>(p, v, n);
  };
  auto gemm = [&](const _Float16* A, const _Float16* Bt, const float* bias, float* Cm,
                  int M, int N, int K, int act) {
    int tiles = (M / 16) * (N / 64);
    k_gemm_wmma<<<(tiles + 3) / 4, 128, 0, stream>>>(A, Bt, bias, Cm, M, N, K, act);
  };
  auto logits = [&](const float* h, const float* a, float* o, int n, int heads, int ch) {
    int t = n * heads;
    k_logits<<<(t + 255) / 256, 256, 0, stream>>>(h, a, o, n, heads, ch);
  };
  auto edge_phase = [&](const float* hsrc, const int* src, const int* dst,
                        int ndst, int heads, int ch, float* acc) {
    fill(mb, NEG_INF, ndst * heads);
    fill(zb, 0.0f, ndst * heads);
    int eh = E * heads;
    k_edge_a<<<(eh + 255) / 256, 256, 0, stream>>>(als, ald, src, dst, eb, mb, E, heads);
    k_edge_b<<<(eh + 255) / 256, 256, 0, stream>>>(eb, mb, dst, zb, E, heads);
    int tot = E * ((heads * ch) >> 2);
    k_edge_c<<<(tot + 255) / 256, 256, 0, stream>>>(hsrc, eb, zb, src, dst, acc, E, heads, ch);
  };

  // ---- weight conversion: f16 + transpose to [N,K] ----
  cvt_t(linW_drug, wld, IND, HID);
  cvt_t(linW_prot, wlp, INP, HID);
  for (int i = 0; i < 6; ++i)
    cvt_t(W_hid + (size_t)i * HID * HID, whid + (size_t)i * HID * HID, HID, HID);
  for (int t = 0; t < 3; ++t)
    cvt_t(W_out + (size_t)t * HID * OUT, wout + (size_t)t * HID * OUT, HID, OUT);
  cvt(x_drug, xdh, ND * IND);
  cvt(x_prot, xph, NP * INP);

  // ---- input projections + relu ----
  gemm(xdh, wld, linb_drug, xd, ND, HID, IND, 1);
  gemm(xph, wlp, linb_prot, xp, NP, HID, INP, 1);
  cvt(xd, xdh, ND * HID);
  cvt(xp, xph, NP * HID);

  // ---- two hidden HeteroConv layers (H=4, C=64, concat) ----
  for (int l = 0; l < 2; ++l) {
    const _Float16* W0 = whid + (size_t)(l * 3 + 0) * HID * HID;
    const _Float16* W1 = whid + (size_t)(l * 3 + 1) * HID * HID;
    const _Float16* W2 = whid + (size_t)(l * 3 + 2) * HID * HID;
    const float* as0 = asrc_hid + (l * 3 + 0) * H * Cc;
    const float* as1 = asrc_hid + (l * 3 + 1) * H * Cc;
    const float* as2 = asrc_hid + (l * 3 + 2) * H * Cc;
    const float* ad0 = adst_hid + (l * 3 + 0) * H * Cc;
    const float* ad1 = adst_hid + (l * 3 + 1) * H * Cc;
    const float* ad2 = adst_hid + (l * 3 + 2) * H * Cc;

    fill(accd, 0.0f, ND * HID);
    fill(accp, 0.0f, NP * HID);

    // conv 0: drug -> protein
    gemm(xdh, W0, nullptr, hd, ND, HID, HID, 0);   // h_src (drug)
    gemm(xph, W0, nullptr, hp, NP, HID, HID, 0);   // h_dst (prot)
    logits(hd, as0, als, ND, H, Cc);
    logits(hp, ad0, ald, NP, H, Cc);
    edge_phase(hd, src_dp, dst_dp, NP, H, Cc, accp);

    // conv 1: protein -> drug
    gemm(xph, W1, nullptr, hp, NP, HID, HID, 0);   // h_src (prot)
    gemm(xdh, W1, nullptr, hd, ND, HID, HID, 0);   // h_dst (drug)
    logits(hp, as1, als, NP, H, Cc);
    logits(hd, ad1, ald, ND, H, Cc);
    edge_phase(hp, src_pd, dst_pd, ND, H, Cc, accd);

    // conv 2: protein -> protein (shared linear: one GEMM)
    gemm(xph, W2, nullptr, hp, NP, HID, HID, 0);
    logits(hp, as2, als, NP, H, Cc);
    logits(hp, ad2, ald, NP, H, Cc);
    edge_phase(hp, src_pp, dst_pp, NP, H, Cc, accp);

    // HeteroConv sum + bias + elu
    int td = ND * HID, tp = NP * HID;
    k_finalize<<<(td + 255) / 256, 256, 0, stream>>>(xd, accd,
        b_hid + (l * 3 + 1) * HID, nullptr, ND, HID, 1);
    k_finalize<<<(tp + 255) / 256, 256, 0, stream>>>(xp, accp,
        b_hid + (l * 3 + 0) * HID, b_hid + (l * 3 + 2) * HID, NP, HID, 1);
    cvt(xd, xdh, ND * HID);
    cvt(xp, xph, NP * HID);
  }

  // ---- output HeteroConv layer (heads=1, OUT=64) ----
  const _Float16* V0 = wout + (size_t)0 * HID * OUT;
  const _Float16* V1 = wout + (size_t)1 * HID * OUT;
  const _Float16* V2 = wout + (size_t)2 * HID * OUT;

  fill(accd, 0.0f, ND * OUT);
  fill(accp, 0.0f, NP * OUT);

  // conv 0: drug -> protein
  gemm(xdh, V0, nullptr, hd, ND, OUT, HID, 0);
  gemm(xph, V0, nullptr, hp, NP, OUT, HID, 0);
  logits(hd, asrc_out + 0 * OUT, als, ND, 1, OUT);
  logits(hp, adst_out + 0 * OUT, ald, NP, 1, OUT);
  edge_phase(hd, src_dp, dst_dp, NP, 1, OUT, accp);

  // conv 1: protein -> drug
  gemm(xph, V1, nullptr, hp, NP, OUT, HID, 0);
  gemm(xdh, V1, nullptr, hd, ND, OUT, HID, 0);
  logits(hp, asrc_out + 1 * OUT, als, NP, 1, OUT);
  logits(hd, adst_out + 1 * OUT, ald, ND, 1, OUT);
  edge_phase(hp, src_pd, dst_pd, ND, 1, OUT, accd);

  // conv 2: protein -> protein
  gemm(xph, V2, nullptr, hp, NP, OUT, HID, 0);
  logits(hp, asrc_out + 2 * OUT, als, NP, 1, OUT);
  logits(hp, adst_out + 2 * OUT, ald, NP, 1, OUT);
  edge_phase(hp, src_pp, dst_pp, NP, 1, OUT, accp);

  // ---- write outputs: od first, then op1+op2 ----
  float* out = (float*)d_out;
  int td = ND * OUT, tp = NP * OUT;
  k_finalize<<<(td + 255) / 256, 256, 0, stream>>>(out, accd,
      b_out + 1 * OUT, nullptr, ND, OUT, 0);
  k_finalize<<<(tp + 255) / 256, 256, 0, stream>>>(out + (size_t)ND * OUT, accp,
      b_out + 0 * OUT, b_out + 2 * OUT, NP, OUT, 0);
}
